// PointNet_31404800868723
// MI455X (gfx1250) — compile-verified, matrix operand order checked by reading, not run on hardware
//
#include <hip/hip_runtime.h>
#include <hip/hip_bf16.h>

#define N_NODES 20000
#define N_EDGES 400000
#define HDIM    256

typedef __attribute__((ext_vector_type(16))) _Float16 v16h;
typedef __attribute__((ext_vector_type(8)))  _Float16 v8h;
typedef __attribute__((ext_vector_type(8)))  float    v8f;

// K index for element i (0..15) of a 16-bit A/B WMMA fragment (CDNA5 layout):
// lanes 0-15 hold K = 0..7,16..23 ; lanes 16-31 hold K = 8..15,24..31
__device__ __forceinline__ int wk(int lane, int i) {
    return (i < 8 ? i : i + 8) + ((lane & 16) ? 8 : 0);
}

// Load a pre-swizzled (fragment-major) operand: 16 contiguous halves -> 2x ds_load_b128
__device__ __forceinline__ v16h ld_frag(const _Float16* p) {
    return *(const v16h*)p;
}

// Build a B-fragment from a row-major f16 row (contiguous K): 2x 16B loads
__device__ __forceinline__ v16h ld_bfrag_f16row(const _Float16* row, int k0, int lane) {
    const int kb = k0 + ((lane & 16) ? 8 : 0);
    v8h lo = *(const v8h*)(row + kb);
    v8h hi = *(const v8h*)(row + kb + 16);
    v16h b;
#pragma unroll
    for (int i = 0; i < 8; ++i) { b[i] = lo[i]; b[i + 8] = hi[i]; }
    return b;
}

// Build a B-fragment from a row-major f32 row (global): 4x global_load_b128 + cvt
__device__ __forceinline__ v16h ld_bfrag_f32row(const float* row, int k0, int lane) {
    const int kb = k0 + ((lane & 16) ? 8 : 0);
    float4 a0 = *(const float4*)(row + kb);
    float4 a1 = *(const float4*)(row + kb + 4);
    float4 b0 = *(const float4*)(row + kb + 16);
    float4 b1 = *(const float4*)(row + kb + 20);
    v16h b;
    b[0]  = (_Float16)a0.x; b[1]  = (_Float16)a0.y; b[2]  = (_Float16)a0.z; b[3]  = (_Float16)a0.w;
    b[4]  = (_Float16)a1.x; b[5]  = (_Float16)a1.y; b[6]  = (_Float16)a1.z; b[7]  = (_Float16)a1.w;
    b[8]  = (_Float16)b0.x; b[9]  = (_Float16)b0.y; b[10] = (_Float16)b0.z; b[11] = (_Float16)b0.w;
    b[12] = (_Float16)b1.x; b[13] = (_Float16)b1.y; b[14] = (_Float16)b1.z; b[15] = (_Float16)b1.w;
    return b;
}

// ---------------- utility kernels ----------------

__global__ __launch_bounds__(256) void k_zero(float* __restrict__ p, long n) {
    for (long i = (long)blockIdx.x * blockDim.x + threadIdx.x; i < n;
         i += (long)gridDim.x * blockDim.x)
        p[i] = 0.0f;
}

// Swizzle W (row-major [Kreal x Ncols] f32, zero-padded to Kpad) into WMMA
// A-fragment-major f16 layout for the TRANSPOSED product:
//   out[((kc*NT + nt)*32 + lane)*16 + i] = W[kc*32 + wk(lane,i)][nt*16 + (lane&15)]
__global__ __launch_bounds__(256) void k_swz(const float* __restrict__ W,
                                             _Float16* __restrict__ out,
                                             int Kreal, int Kpad, int Ncols) {
    const int NT = Ncols / 16;
    const int total = (Kpad / 32) * NT * 512;
    for (int idx = blockIdx.x * blockDim.x + threadIdx.x; idx < total;
         idx += gridDim.x * blockDim.x) {
        int i    = idx & 15;
        int lane = (idx >> 4) & 31;
        int t    = idx >> 9;
        int nt   = t % NT;
        int kc   = t / NT;
        int k    = kc * 32 + wk(lane, i);
        int n    = nt * 16 + (lane & 15);
        out[idx] = (k < Kreal) ? (_Float16)W[k * Ncols + n] : (_Float16)0.0f;
    }
}

// ---------------- PointNet edge-MLP + fused segment_max (the hot kernel) ----------------
// Transposed form: D = W_tileT (A operand, pre-swizzled) x activationsT (B operand).
// 2-way edge-tile blocking: each wave owns 32 edges -> every weight fragment feeds
// two independent WMMA chains (halves LDS weight traffic, hides WMMA hazards).
__global__ __launch_bounds__(256) void k_pointnet(const float* __restrict__ pos,
                                                  const int* __restrict__ ei,
                                                  const _Float16* __restrict__ gW1f,
                                                  const _Float16* __restrict__ gW2f,
                                                  const float* __restrict__ b1,
                                                  const float* __restrict__ b2,
                                                  float* __restrict__ hout, int hbase) {
    extern __shared__ char smem[];
    _Float16* sW1  = (_Float16*)smem;                 // 1*16*512   = 8192  halves
    _Float16* sW2  = sW1 + 16 * 512;                  // 8*16*512   = 65536 halves
    float*    sb1  = (float*)(sW2 + 8 * 16 * 512);    // 256
    float*    sb2  = sb1 + HDIM;                      // 256
    _Float16* sHid = (_Float16*)(sb2 + HDIM);         // 8 waves * 32 * 256 halves

    const int tid = threadIdx.x;
    for (int i = tid; i < 16 * 512; i += blockDim.x) sW1[i] = gW1f[i];
    for (int i = tid; i < 8 * 16 * 512; i += blockDim.x) sW2[i] = gW2f[i];
    for (int i = tid; i < HDIM; i += blockDim.x) { sb1[i] = b1[i]; sb2[i] = b2[i]; }
    __syncthreads();

    const int lane  = tid & 31;
    const int wave  = tid >> 5;
    const int n     = lane & 15;              // edge slot within tile
    const int hioff = (lane & 16) ? 8 : 0;    // output-column half select
    _Float16* hid = sHid + wave * 32 * HDIM;  // rows 0..15 tile0, 16..31 tile1

    const int nst = N_EDGES / 32;             // 32-edge supertiles
    const int ststride = gridDim.x * 8;
    for (int st = blockIdx.x * 8 + wave; st < nst; st += ststride) {
        const int e0 = st * 32;
        const int eA = e0 + n;
        const int eB = e0 + 16 + n;
        // prefetch next supertile's edge indices (global_prefetch_b8)
        {
            const int en = e0 + ststride * 32;
            if (en < N_EDGES) {
                __builtin_prefetch(&ei[en + n], 0, 0);
                __builtin_prefetch(&ei[N_EDGES + en + n], 0, 0);
            }
        }
        const int sA = ei[eA], dA = ei[N_EDGES + eA];
        const int sB = ei[eB], dB = ei[N_EDGES + eB];
        // ---- feature B-fragments: [32 K x 16 edges] x2, 6 real features each
        float fA[6], fB[6];
        {
            float sx = pos[3 * sA], sy = pos[3 * sA + 1], sz = pos[3 * sA + 2];
            fA[0] = sx; fA[1] = sy; fA[2] = sz;
            fA[3] = sx - pos[3 * dA];
            fA[4] = sy - pos[3 * dA + 1];
            fA[5] = sz - pos[3 * dA + 2];
            float tx = pos[3 * sB], ty = pos[3 * sB + 1], tz = pos[3 * sB + 2];
            fB[0] = tx; fB[1] = ty; fB[2] = tz;
            fB[3] = tx - pos[3 * dB];
            fB[4] = ty - pos[3 * dB + 1];
            fB[5] = tz - pos[3 * dB + 2];
        }
        v16h bf0, bf1;
#pragma unroll
        for (int i = 0; i < 16; ++i) {
            int k = wk(lane, i);
            bf0[i] = (k < 6) ? (_Float16)fA[k] : (_Float16)0.0f;
            bf1[i] = (k < 6) ? (_Float16)fB[k] : (_Float16)0.0f;
        }
        // ---- hiddenT = relu(W1T @ featT + b1): one b128 store per lane per tile
#pragma unroll 1
        for (int nt = 0; nt < 16; ++nt) {
            v16h af = ld_frag(sW1 + (nt * 32 + lane) * 16);
            v8f acc0 = {}, acc1 = {};
            acc0 = __builtin_amdgcn_wmma_f32_16x16x32_f16(false, af, false, bf0,
                                                          (short)0, acc0, false, false);
            acc1 = __builtin_amdgcn_wmma_f32_16x16x32_f16(false, af, false, bf1,
                                                          (short)0, acc1, false, false);
            v8h h0, h1;
#pragma unroll
            for (int r = 0; r < 8; ++r) {
                float bb = sb1[nt * 16 + hioff + r];
                float v0 = acc0[r] + bb;
                float v1 = acc1[r] + bb;
                h0[r] = (_Float16)(v0 > 0.0f ? v0 : 0.0f);
                h1[r] = (_Float16)(v1 > 0.0f ? v1 : 0.0f);
            }
            *(v8h*)(hid + n * HDIM + nt * 16 + hioff) = h0;          // ds_store_b128
            *(v8h*)(hid + (16 + n) * HDIM + nt * 16 + hioff) = h1;   // ds_store_b128
        }
        // ---- mT = W2T @ hiddenT + b2 ; atomic-max scatter (dst init 0 == fused relu)
        float* outA = &hout[(long)dA * (2 * HDIM) + hbase];
        float* outB = &hout[(long)dB * (2 * HDIM) + hbase];
#pragma unroll 1
        for (int nt = 0; nt < 16; ++nt) {
            v8f acc0 = {}, acc1 = {};
#pragma unroll
            for (int kc = 0; kc < 8; ++kc) {
                v16h af  = ld_frag(sW2 + ((kc * 16 + nt) * 32 + lane) * 16);
                v16h bh0 = ld_bfrag_f16row(hid + n * HDIM, kc * 32, lane);
                v16h bh1 = ld_bfrag_f16row(hid + (16 + n) * HDIM, kc * 32, lane);
                acc0 = __builtin_amdgcn_wmma_f32_16x16x32_f16(false, af, false, bh0,
                                                              (short)0, acc0, false, false);
                acc1 = __builtin_amdgcn_wmma_f32_16x16x32_f16(false, af, false, bh1,
                                                              (short)0, acc1, false, false);
            }
#pragma unroll
            for (int r = 0; r < 8; ++r) {
                int col = nt * 16 + hioff + r;
                float bb = sb2[col];
                // signed-int max == float max when one side >= 0 (dst init 0)
                atomicMax((int*)&outA[col], __float_as_int(acc0[r] + bb));
                atomicMax((int*)&outB[col], __float_as_int(acc1[r] + bb));
            }
        }
    }
}

// ---------------- generic WMMA GEMM (transposed): C = opt_relu(A@B + bias) ----------------
// A: f32 row-major [M x K] global; Bf: pre-swizzled f16 fragments of B (K x Ncols)
// 2-way row-tile blocking: each job covers 32 rows, reusing every weight fragment twice.
__global__ __launch_bounds__(256) void k_gemm(const float* __restrict__ A,
                                              const _Float16* __restrict__ Bf,
                                              const float* __restrict__ bias,
                                              float* __restrict__ C,
                                              int M, int K, int Ncols,
                                              int ldc, int cbase, int relu) {
    extern __shared__ char smem[];
    _Float16* sB    = (_Float16*)smem;          // K*Ncols halves (fragment-major)
    float*    sbias = (float*)(sB + K * Ncols); // Ncols

    const int tid = threadIdx.x;
    for (int i = tid; i < K * Ncols; i += blockDim.x) sB[i] = Bf[i];
    for (int i = tid; i < Ncols; i += blockDim.x) sbias[i] = bias ? bias[i] : 0.0f;
    __syncthreads();

    const int lane  = tid & 31;
    const int wave  = tid >> 5;
    const int n     = lane & 15;               // row (node) within tile
    const int hioff = (lane & 16) ? 8 : 0;
    const int waves = blockDim.x >> 5;

    const int MT2 = M / 32, NT = Ncols / 16;
    const int jobs = MT2 * NT;
    for (int job = blockIdx.x * waves + wave; job < jobs; job += gridDim.x * waves) {
        const int mt = job / NT, nt = job % NT;
        const int row0 = mt * 32 + n;
        const int row1 = mt * 32 + 16 + n;
        const float* ap0 = A + (long)row0 * K;
        const float* ap1 = A + (long)row1 * K;
        v8f acc0 = {}, acc1 = {};
#pragma unroll 1
        for (int k0 = 0; k0 < K; k0 += 32) {
            v16h af  = ld_frag(sB + (((k0 >> 5) * NT + nt) * 32 + lane) * 16);
            v16h bf0 = ld_bfrag_f32row(ap0, k0, lane);
            v16h bf1 = ld_bfrag_f32row(ap1, k0, lane);
            acc0 = __builtin_amdgcn_wmma_f32_16x16x32_f16(false, af, false, bf0,
                                                          (short)0, acc0, false, false);
            acc1 = __builtin_amdgcn_wmma_f32_16x16x32_f16(false, af, false, bf1,
                                                          (short)0, acc1, false, false);
        }
        float4 o[4];
#pragma unroll
        for (int half = 0; half < 2; ++half) {
            const v8f& acc = half ? acc1 : acc0;
#pragma unroll
            for (int q = 0; q < 2; ++q) {
                float4 t;
                float* tp = &t.x;
#pragma unroll
                for (int j = 0; j < 4; ++j) {
                    int r = q * 4 + j;
                    float v = acc[r] + sbias[nt * 16 + hioff + r];
                    if (relu) v = v > 0.0f ? v : 0.0f;
                    tp[j] = v;
                }
                o[half * 2 + q] = t;
            }
        }
        float* c0 = &C[(long)row0 * ldc + cbase + nt * 16 + hioff];
        float* c1 = &C[(long)row1 * ldc + cbase + nt * 16 + hioff];
        *(float4*)(c0)     = o[0];
        *(float4*)(c0 + 4) = o[1];
        *(float4*)(c1)     = o[2];
        *(float4*)(c1 + 4) = o[3];
    }
}

// ---------------- GCN support kernels ----------------

__global__ __launch_bounds__(256) void k_deg_init(float* __restrict__ deg) {
    int n = blockIdx.x * blockDim.x + threadIdx.x;
    if (n < N_NODES) deg[n] = 1.0f;
}

__global__ __launch_bounds__(256) void k_deg_acc(const int* __restrict__ ei,
                                                 float* __restrict__ deg) {
    int e = blockIdx.x * blockDim.x + threadIdx.x;
    if (e < N_EDGES) atomicAdd(&deg[ei[N_EDGES + e]], 1.0f);
}

__global__ __launch_bounds__(256) void k_dinv(float* __restrict__ deg) {
    int n = blockIdx.x * blockDim.x + threadIdx.x;
    if (n < N_NODES) deg[n] = rsqrtf(deg[n]);
}

// agg[dst] += xw[src] * dinv[src]*dinv[dst] ; 32 lanes x 4 cols per edge
__global__ __launch_bounds__(256) void k_agg(const float* __restrict__ xw,
                                             const int* __restrict__ ei,
                                             const float* __restrict__ dinv,
                                             float* __restrict__ agg) {
    const long total = (long)N_EDGES * 32;
    for (long t = (long)blockIdx.x * blockDim.x + threadIdx.x; t < total;
         t += (long)gridDim.x * blockDim.x) {
        int e  = (int)(t >> 5);
        int c0 = (int)(t & 31) * 4;
        int s = ei[e], d = ei[N_EDGES + e];
        float nrm = dinv[s] * dinv[d];
        const float4 v = *(const float4*)&xw[(long)s * 128 + c0];
        atomicAdd(&agg[(long)d * 128 + c0 + 0], v.x * nrm);
        atomicAdd(&agg[(long)d * 128 + c0 + 1], v.y * nrm);
        atomicAdd(&agg[(long)d * 128 + c0 + 2], v.z * nrm);
        atomicAdd(&agg[(long)d * 128 + c0 + 3], v.w * nrm);
    }
}

// h2[:, base:base+128] = relu(agg + xw * dinv^2 + b)
__global__ __launch_bounds__(256) void k_self_bias(const float* __restrict__ agg,
                                                   const float* __restrict__ xw,
                                                   const float* __restrict__ dinv,
                                                   const float* __restrict__ b,
                                                   float* __restrict__ h2, int base) {
    const long total = (long)N_NODES * 128;
    for (long t = (long)blockIdx.x * blockDim.x + threadIdx.x; t < total;
         t += (long)gridDim.x * blockDim.x) {
        int n = (int)(t >> 7), c = (int)(t & 127);
        float di = dinv[n];
        float v = agg[t] + xw[t] * di * di + b[c];
        h2[(long)n * 256 + base + c] = v > 0.0f ? v : 0.0f;
    }
}

// ---------------- classifier ----------------

__global__ void k_cls_init(const float* __restrict__ cls_b, float* __restrict__ out) {
    int j = threadIdx.x;
    if (j < 3) out[j] = cls_b[j];
}

__global__ __launch_bounds__(256) void k_cls(const float* __restrict__ h3,
                                             const float* __restrict__ W,
                                             float* __restrict__ out) {
    float s0 = 0.f, s1 = 0.f, s2 = 0.f;
    const long total = (long)N_NODES * 128;
    for (long i = (long)blockIdx.x * blockDim.x + threadIdx.x; i < total;
         i += (long)gridDim.x * blockDim.x) {
        float v = h3[i];
        s0 += v * W[i * 3 + 0];
        s1 += v * W[i * 3 + 1];
        s2 += v * W[i * 3 + 2];
    }
    __shared__ float red[3][256];
    int tid = threadIdx.x;
    red[0][tid] = s0; red[1][tid] = s1; red[2][tid] = s2;
    __syncthreads();
    for (int off = 128; off > 0; off >>= 1) {
        if (tid < off) {
            red[0][tid] += red[0][tid + off];
            red[1][tid] += red[1][tid + off];
            red[2][tid] += red[2][tid + off];
        }
        __syncthreads();
    }
    if (tid == 0) {
        atomicAdd(&out[0], red[0][0]);
        atomicAdd(&out[1], red[1][0]);
        atomicAdd(&out[2], red[2][0]);
    }
}

// ---------------- host launch ----------------

extern "C" void kernel_launch(void* const* d_in, const int* in_sizes, int n_in,
                              void* d_out, int out_size, void* d_ws, size_t ws_size,
                              hipStream_t stream) {
    const float* pos0  = (const float*)d_in[0];
    const float* pos1  = (const float*)d_in[1];
    const int*   ei0   = (const int*)d_in[2];
    const int*   ei1   = (const int*)d_in[3];
    const float* p0_W1 = (const float*)d_in[6];
    const float* p0_b1 = (const float*)d_in[7];
    const float* p0_W2 = (const float*)d_in[8];
    const float* p0_b2 = (const float*)d_in[9];
    const float* p1_W1 = (const float*)d_in[10];
    const float* p1_b1 = (const float*)d_in[11];
    const float* p1_W2 = (const float*)d_in[12];
    const float* p1_b2 = (const float*)d_in[13];
    const float* g0_W  = (const float*)d_in[14];
    const float* g0_b  = (const float*)d_in[15];
    const float* g1_W  = (const float*)d_in[16];
    const float* g1_b  = (const float*)d_in[17];
    const float* c3_W  = (const float*)d_in[18];
    const float* c3_b  = (const float*)d_in[19];
    const float* cls_W = (const float*)d_in[20];
    const float* cls_b = (const float*)d_in[21];
    float* out = (float*)d_out;

    // workspace carve-up
    float* ws   = (float*)d_ws;
    float* h    = ws;                                // [N, 512] pointnet concat
    float* xw0  = h    + (size_t)N_NODES * 512;      // [N, 128]
    float* xw1  = xw0  + (size_t)N_NODES * 128;
    float* agg0 = xw1  + (size_t)N_NODES * 128;
    float* agg1 = agg0 + (size_t)N_NODES * 128;
    float* h2   = agg1 + (size_t)N_NODES * 128;      // [N, 256] gcn concat
    float* h3   = h2   + (size_t)N_NODES * 256;      // [N, 128]
    float* deg0 = h3   + (size_t)N_NODES * 128;      // [N]
    float* deg1 = deg0 + N_NODES;
    _Float16* fp    = (_Float16*)(deg1 + N_NODES);
    _Float16* p0W1f = fp; fp += 16 * 512;            // (32/32)*(256/16)*512
    _Float16* p0W2f = fp; fp += 8 * 16 * 512;        // (256/32)*(256/16)*512
    _Float16* p1W1f = fp; fp += 16 * 512;
    _Float16* p1W2f = fp; fp += 8 * 16 * 512;
    _Float16* g0Wf  = fp; fp += 16 * 8 * 512;        // (512/32)*(128/16)*512
    _Float16* g1Wf  = fp; fp += 16 * 8 * 512;
    _Float16* c3Wf  = fp; fp += 8 * 8 * 512;         // (256/32)*(128/16)*512

    // init accumulators
    k_zero<<<4096, 256, 0, stream>>>(h, (long)N_NODES * 512);
    k_zero<<<2048, 256, 0, stream>>>(agg0, (long)N_NODES * 128);
    k_zero<<<2048, 256, 0, stream>>>(agg1, (long)N_NODES * 128);

    // weight swizzle to WMMA fragment-major f16
    k_swz<<<64, 256, 0, stream>>>(p0_W1, p0W1f, 6, 32, HDIM);
    k_swz<<<64, 256, 0, stream>>>(p1_W1, p1W1f, 6, 32, HDIM);
    k_swz<<<256, 256, 0, stream>>>(p0_W2, p0W2f, HDIM, HDIM, HDIM);
    k_swz<<<256, 256, 0, stream>>>(p1_W2, p1W2f, HDIM, HDIM, HDIM);
    k_swz<<<256, 256, 0, stream>>>(g0_W, g0Wf, 512, 512, 128);
    k_swz<<<256, 256, 0, stream>>>(g1_W, g1Wf, 512, 512, 128);
    k_swz<<<128, 256, 0, stream>>>(c3_W, c3Wf, 256, 256, 128);

    // PointNet layers (fused edge-MLP + segment_max + relu)
    const size_t pn_lds = (size_t)(16 * 512 + 8 * 16 * 512 + 8 * 32 * HDIM) * 2
                        + 2 * HDIM * 4;              // ~274 KB
    const int pn_blocks = (N_EDGES / 32 + 7) / 8;    // 1563
    k_pointnet<<<pn_blocks, 256, pn_lds, stream>>>(pos0, ei0, p0W1f, p0W2f,
                                                   p0_b1, p0_b2, h, 0);
    k_pointnet<<<pn_blocks, 256, pn_lds, stream>>>(pos1, ei1, p1W1f, p1W2f,
                                                   p1_b1, p1_b2, h, HDIM);

    // degrees -> dinv
    k_deg_init<<<(N_NODES + 255) / 256, 256, 0, stream>>>(deg0);
    k_deg_init<<<(N_NODES + 255) / 256, 256, 0, stream>>>(deg1);
    k_deg_acc<<<(N_EDGES + 255) / 256, 256, 0, stream>>>(ei0, deg0);
    k_deg_acc<<<(N_EDGES + 255) / 256, 256, 0, stream>>>(ei1, deg1);
    k_dinv<<<(N_NODES + 255) / 256, 256, 0, stream>>>(deg0);
    k_dinv<<<(N_NODES + 255) / 256, 256, 0, stream>>>(deg1);

    // xw = h @ gW  (WMMA)
    const size_t g_lds = (size_t)512 * 128 * 2 + 128 * 4;
    const int g_jobs = (N_NODES / 32) * (128 / 16);  // 5000
    k_gemm<<<(g_jobs + 7) / 8, 256, g_lds, stream>>>(h, g0Wf, nullptr, xw0,
                                                     N_NODES, 512, 128, 128, 0, 0);
    k_gemm<<<(g_jobs + 7) / 8, 256, g_lds, stream>>>(h, g1Wf, nullptr, xw1,
                                                     N_NODES, 512, 128, 128, 0, 0);

    // neighborhood aggregation (segment_sum) + self loop + bias + relu
    k_agg<<<4096, 256, 0, stream>>>(xw0, ei0, deg0, agg0);
    k_agg<<<4096, 256, 0, stream>>>(xw1, ei1, deg1, agg1);
    k_self_bias<<<4096, 256, 0, stream>>>(agg0, xw0, deg0, g0_b, h2, 0);
    k_self_bias<<<4096, 256, 0, stream>>>(agg1, xw1, deg1, g1_b, h2, 128);

    // conv3: h3 = relu(h2 @ c3_W + c3_b)  (WMMA)
    const size_t c_lds = (size_t)256 * 128 * 2 + 128 * 4;
    k_gemm<<<(g_jobs + 7) / 8, 256, c_lds, stream>>>(h2, c3Wf, c3_b, h3,
                                                     N_NODES, 256, 128, 128, 0, 1);

    // classifier: out = h3.flatten() @ cls_W + cls_b
    k_cls_init<<<1, 4, 0, stream>>>(cls_b, out);
    k_cls<<<2048, 256, 0, stream>>>(h3, cls_W, out);
}
